// _GINBackbone_36421322670670
// MI455X (gfx1250) — matrix operand order repaired and checked by reading, not compiled, not run on hardware
//
#include <hip/hip_runtime.h>
#include <hip/hip_bf16.h>

typedef __attribute__((ext_vector_type(16))) _Float16 v16h;
typedef __attribute__((ext_vector_type(8)))  float    v8f;

#define DFEAT 64
#define GEMM_TILES 4   // 16-row tiles per wave

// ---------------------------------------------------------------- utilities
__global__ void k_zero_f32(float* __restrict__ p, int n) {
    int i = blockIdx.x * blockDim.x + threadIdx.x;
    if (i < n) p[i] = 0.0f;
}

__global__ void k_copy_f4(const float* __restrict__ src, float* __restrict__ dst, int n4) {
    int i = blockIdx.x * blockDim.x + threadIdx.x;
    if (i < n4) ((float4*)dst)[i] = ((const float4*)src)[i];
}

// Pre-pack W1 [3,64,64] and W2 [3,64,64] (fp32, row-major [k][n]) into fp16
// WMMA B-fragment lane order:
//   packed[mat][ (s*4+t)*32 + lane ][i]  =  W[k][n]
//   with k = s*32 + (i>>3)*16 + (lane>>4)*8 + (i&7),  n = t*16 + (lane&15)
// so a lane's whole fragment is one contiguous 32-byte (v16h) load.
__global__ void k_pack_w(const float* __restrict__ W1, const float* __restrict__ W2,
                         _Float16* __restrict__ wp) {
    int j = blockIdx.x * blockDim.x + threadIdx.x;      // 6*4096
    if (j >= 6 * 4096) return;
    int m    = j >> 12;         // matrix 0..5 (0..2 = W1 layers, 3..5 = W2 layers)
    int r    = j & 4095;
    int s    = r >> 11;         // k-step 0..1
    int t    = (r >> 9) & 3;    // column tile 0..3
    int lane = (r >> 4) & 31;
    int i    = r & 15;
    int k = s * 32 + (i >> 3) * 16 + (lane >> 4) * 8 + (i & 7);
    int n = t * 16 + (lane & 15);
    const float* Wsrc = (m < 3) ? (W1 + (size_t)m * 4096) : (W2 + (size_t)(m - 3) * 4096);
    wp[j] = (_Float16)Wsrc[k * DFEAT + n];
}

// ---------------------------------------------------------------- edge scatter
// z[dst[e]] += h[src[e]]   (GIN sum aggregation; z pre-initialized to h -> z = h + agg)
__global__ void k_edge_scatter(const float* __restrict__ h,
                               const int* __restrict__ src, const int* __restrict__ dst,
                               float* __restrict__ z, int E) {
    int idx = blockIdx.x * blockDim.x + threadIdx.x;
    int e = idx >> 4;
    if (e >= E) return;
    int c4 = (idx & 15) << 2;
    int s = src[e], d = dst[e];
    float4 v = *(const float4*)(h + (size_t)s * DFEAT + c4);
    float* p = z + (size_t)d * DFEAT + c4;
    unsafeAtomicAdd(p + 0, v.x);
    unsafeAtomicAdd(p + 1, v.y);
    unsafeAtomicAdd(p + 2, v.z);
    unsafeAtomicAdd(p + 3, v.w);
}

// ---------------------------------------------------------------- fused WMMA GEMM
// Y[row,:] = act(A[row,:]) @ W + bias ; act = relu(a*x+b) iff BN (compile-time).
// One wave holds all 8 packed B fragments in registers and sweeps GEMM_TILES
// 16-row tiles. Accumulators start at zero (inline-0 SRC2); bias is added in
// the f32 epilogue together with per-column sum/sumsq accumulation for the
// following BatchNorm (lane-pair reduced -> one atomic per column per tile).
template <bool BN>
__global__ __launch_bounds__(32) void k_gin_gemm(
        const float*    __restrict__ A,      // [nRows,64]
        const _Float16* __restrict__ Wp,     // packed [8][32] v16h fragments
        const float*    __restrict__ bias,   // [64]
        const float*    __restrict__ bnAB,   // [128] = a[64],b[64] (used iff BN)
        float*          __restrict__ Y,      // [nRows,64]
        float* __restrict__ colSum, float* __restrict__ colSq,
        int nTiles) {
    const int lane = threadIdx.x;
    const int grp  = lane >> 4;      // 0 | 1
    const int lm   = lane & 15;

    // all 8 B fragments: [s][t], one v16h (2x b128) load each
    const v16h* Bp = (const v16h*)Wp;
    v16h bf[2][4];
#pragma unroll
    for (int s = 0; s < 2; ++s)
#pragma unroll
        for (int t = 0; t < 4; ++t)
            bf[s][t] = Bp[(s * 4 + t) * 32 + lane];

    float bv[4];
#pragma unroll
    for (int t = 0; t < 4; ++t) bv[t] = bias[t * 16 + lm];

    // BN coefficients for this lane's A elements (k depends only on lane group)
    float bna[2][16], bnb[2][16];
    if (BN) {
#pragma unroll
        for (int s = 0; s < 2; ++s)
#pragma unroll
            for (int i = 0; i < 8; ++i) {
                int k0 = s * 32 + grp * 8 + i;
                int k1 = s * 32 + 16 + grp * 8 + i;
                bna[s][i]     = bnAB[k0];  bnb[s][i]     = bnAB[64 + k0];
                bna[s][i + 8] = bnAB[k1];  bnb[s][i + 8] = bnAB[64 + k1];
            }
    }

    for (int j = 0; j < GEMM_TILES; ++j) {
        const int tile = blockIdx.x * GEMM_TILES + j;
        if (tile >= nTiles) break;                 // uniform across the wave
        const int m0 = tile * 16;
        const float4* ar = (const float4*)(A + (size_t)(m0 + lm) * DFEAT);

        v8f acc[4] = {v8f{}, v8f{}, v8f{}, v8f{}};

#pragma unroll
        for (int s = 0; s < 2; ++s) {
            // A fragment: halves 0..7 -> K = s*32+grp*8+i ; halves 8..15 -> +16
            float4 p0 = ar[s * 8 + grp * 2 + 0];
            float4 p1 = ar[s * 8 + grp * 2 + 1];
            float4 p2 = ar[s * 8 + 4 + grp * 2 + 0];
            float4 p3 = ar[s * 8 + 4 + grp * 2 + 1];
            float lo[8] = {p0.x, p0.y, p0.z, p0.w, p1.x, p1.y, p1.z, p1.w};
            float hi[8] = {p2.x, p2.y, p2.z, p2.w, p3.x, p3.y, p3.z, p3.w};
            v16h af;
#pragma unroll
            for (int i = 0; i < 8; ++i) {
                float v0 = lo[i], v1 = hi[i];
                if (BN) {
                    v0 = fmaxf(fmaf(v0, bna[s][i],     bnb[s][i]),     0.0f);
                    v1 = fmaxf(fmaf(v1, bna[s][i + 8], bnb[s][i + 8]), 0.0f);
                }
                af[i]     = (_Float16)v0;
                af[i + 8] = (_Float16)v1;
            }
#pragma unroll
            for (int t = 0; t < 4; ++t)
                acc[t] = __builtin_amdgcn_wmma_f32_16x16x32_f16(
                    false, af, false, bf[s][t], (short)0, acc[t], false, false);
        }

        // epilogue: +bias, store, per-column BN statistics
#pragma unroll
        for (int t = 0; t < 4; ++t) {
            const int n = t * 16 + lm;
            float s = 0.0f, q = 0.0f;
#pragma unroll
            for (int r = 0; r < 8; ++r) {
                float v = acc[t][r] + bv[t];
                int m = m0 + grp * 8 + r;
                Y[(size_t)m * DFEAT + n] = v;
                s += v;
                q += v * v;
            }
            // combine the two lanes holding the same column (lane, lane^16)
            s += __shfl_xor(s, 16, 32);
            q += __shfl_xor(q, 16, 32);
            if (grp == 0) {
                unsafeAtomicAdd(colSum + n, s);
                unsafeAtomicAdd(colSq  + n, q);
            }
        }
    }
}

// ---------------------------------------------------------------- BN finalize
__global__ void k_bn_finalize(const float* __restrict__ sum, const float* __restrict__ sq,
                              const float* __restrict__ gamma, const float* __restrict__ beta,
                              float* __restrict__ ab, float invN) {
    int c = threadIdx.x;               // blockDim = 64
    float mean = sum[c] * invN;
    float var  = sq[c] * invN - mean * mean;
    float a    = gamma[c] * rsqrtf(var + 1e-5f);
    ab[c]      = a;
    ab[64 + c] = beta[c] - mean * a;
}

// ---------------------------------------------------------------- outer BN + ReLU + global_add_pool
__global__ void k_bn_pool(const float* __restrict__ y, const float* __restrict__ ab,
                          const int* __restrict__ batch,
                          float* __restrict__ h, float* __restrict__ out,
                          int layer, int nThreads) {
    int idx = blockIdx.x * blockDim.x + threadIdx.x;
    if (idx >= nThreads) return;
    int i  = idx >> 4;
    int c4 = (idx & 15) << 2;
    float4 v = *(const float4*)(y + (size_t)i * DFEAT + c4);
    float4 r;
    r.x = fmaxf(fmaf(v.x, ab[c4 + 0], ab[64 + c4 + 0]), 0.0f);
    r.y = fmaxf(fmaf(v.y, ab[c4 + 1], ab[64 + c4 + 1]), 0.0f);
    r.z = fmaxf(fmaf(v.z, ab[c4 + 2], ab[64 + c4 + 2]), 0.0f);
    r.w = fmaxf(fmaf(v.w, ab[c4 + 3], ab[64 + c4 + 3]), 0.0f);
    *(float4*)(h + (size_t)i * DFEAT + c4) = r;
    float* o = out + (size_t)batch[i] * (3 * DFEAT) + layer * DFEAT + c4;
    unsafeAtomicAdd(o + 0, r.x);
    unsafeAtomicAdd(o + 1, r.y);
    unsafeAtomicAdd(o + 2, r.z);
    unsafeAtomicAdd(o + 3, r.w);
}

// ---------------------------------------------------------------- launch
extern "C" void kernel_launch(void* const* d_in, const int* in_sizes, int n_in,
                              void* d_out, int out_size, void* d_ws, size_t ws_size,
                              hipStream_t stream) {
    const float* x     = (const float*)d_in[0];
    const int*   ei    = (const int*)d_in[1];
    const int*   batch = (const int*)d_in[2];
    const float* W1    = (const float*)d_in[3];
    const float* b1    = (const float*)d_in[4];
    const float* g1    = (const float*)d_in[5];
    const float* bt1   = (const float*)d_in[6];
    const float* W2    = (const float*)d_in[7];
    const float* b2    = (const float*)d_in[8];
    const float* g_out = (const float*)d_in[9];
    const float* b_out = (const float*)d_in[10];
    float* out = (float*)d_out;

    const int N = in_sizes[0] / DFEAT;      // 50000 (multiple of 16)
    const int E = in_sizes[1] / 2;          // 800000
    const int* src = ei;
    const int* dst = ei + E;
    const int L = 3;

    // workspace layout
    float* h     = (float*)d_ws;
    float* z     = h  + (size_t)N * DFEAT;
    float* y1    = z  + (size_t)N * DFEAT;
    float* y2    = y1 + (size_t)N * DFEAT;
    float* stats = y2 + (size_t)N * DFEAT;  // sum1 sq1 sum2 sq2 (256) + ab1(128) + ab2(128)
    _Float16* wp = (_Float16*)(stats + 512); // 6 packed 64x64 f16 matrices

    const int n4      = N * 16;
    const int B256    = 256;
    const int nTiles  = N / 16;                               // 3125
    const int gemmGrd = (nTiles + GEMM_TILES - 1) / GEMM_TILES;

    k_zero_f32<<<(out_size + B256 - 1) / B256, B256, 0, stream>>>(out, out_size);
    k_pack_w  <<<(6 * 4096 + B256 - 1) / B256, B256, 0, stream>>>(W1, W2, wp);
    k_copy_f4 <<<(n4 + B256 - 1) / B256, B256, 0, stream>>>(x, h, n4);

    const float invN = 1.0f / (float)N;
    for (int l = 0; l < L; ++l) {
        k_zero_f32<<<1, 256, 0, stream>>>(stats, 256);
        // z = h (self term), then z += segment_sum(h[src] -> dst)
        k_copy_f4<<<(n4 + B256 - 1) / B256, B256, 0, stream>>>(h, z, n4);
        k_edge_scatter<<<(E * 16 + B256 - 1) / B256, B256, 0, stream>>>(h, src, dst, z, E);
        // y1 = z @ W1 + b1  (+ column stats)
        k_gin_gemm<false><<<gemmGrd, 32, 0, stream>>>(
            z, wp + (size_t)l * 4096, b1 + l * DFEAT, nullptr,
            y1, stats + 0, stats + 64, nTiles);
        k_bn_finalize<<<1, 64, 0, stream>>>(stats + 0, stats + 64,
                                            g1 + l * DFEAT, bt1 + l * DFEAT,
                                            stats + 256, invN);
        // y2 = relu(bn1(y1)) @ W2 + b2  (+ column stats)
        k_gin_gemm<true><<<gemmGrd, 32, 0, stream>>>(
            y1, wp + (size_t)(L + l) * 4096, b2 + l * DFEAT, stats + 256,
            y2, stats + 128, stats + 192, nTiles);
        k_bn_finalize<<<1, 64, 0, stream>>>(stats + 128, stats + 192,
                                            g_out + l * DFEAT, b_out + l * DFEAT,
                                            stats + 384, invN);
        // h = relu(bn_out(y2)); pooled[l] += per-graph sums
        k_bn_pool<<<(n4 + B256 - 1) / B256, B256, 0, stream>>>(y2, stats + 384, batch,
                                                               h, out, l, n4);
    }
}
